// MultiHeadAttention_61134564491715
// MI455X (gfx1250) — matrix operand order repaired and checked by reading, not compiled, not run on hardware
//
#include <hip/hip_runtime.h>
#include <hip/hip_bf16.h>

// MultiHeadAttention for MI455X (gfx1250), bf16 WMMA + flash-attention.
// B=2, S=2048, D=1024, H=16, DK=64. Workspace use: 48 MB.

#define B_  2
#define S_  2048
#define D_  1024
#define H_  16
#define DK_ 64
#define M_  (B_ * S_)

typedef __attribute__((ext_vector_type(16))) __bf16 v16bf;
typedef __attribute__((ext_vector_type(8)))  float  v8f;

union Frag {
  v16bf v;
  uint4 q[2];
};

__device__ __forceinline__ __bf16 f2bf(float f) {
  union { float f; unsigned u; } in;
  in.f = f;
  unsigned u = in.u;
  u += 0x7fffu + ((u >> 16) & 1u);   // round-to-nearest-even
  union { unsigned short s; __bf16 b; } out;
  out.s = (unsigned short)(u >> 16);
  return out.b;
}

__device__ __forceinline__ v8f wmma_bf16(v16bf a, v16bf b, v8f c) {
  // (neg_a, A, neg_b, B, c_mod, C, reuse_a, reuse_b)
  return __builtin_amdgcn_wmma_f32_16x16x32_bf16(false, a, false, b, (short)0, c,
                                                 false, false);
}

// Max/sum across the 16-lane half of a wave32 (rows of one C half live there).
__device__ __forceinline__ float halfmax(float t) {
  t = fmaxf(t, __shfl_xor(t, 1, 32));
  t = fmaxf(t, __shfl_xor(t, 2, 32));
  t = fmaxf(t, __shfl_xor(t, 4, 32));
  t = fmaxf(t, __shfl_xor(t, 8, 32));
  return t;
}
__device__ __forceinline__ float halfsum(float t) {
  t += __shfl_xor(t, 1, 32);
  t += __shfl_xor(t, 2, 32);
  t += __shfl_xor(t, 4, 32);
  t += __shfl_xor(t, 8, 32);
  return t;
}

// ---------------------------------------------------------------- converts
__global__ void cvt_f32_bf16(const float* __restrict__ in, __bf16* __restrict__ out,
                             int n) {
  int i = blockIdx.x * blockDim.x + threadIdx.x;
  if (i < n) out[i] = f2bf(in[i]);
}

// WT[n][k] = W[k][n], f32 -> bf16. Block (32,8), tiles 32x32 via LDS.
__global__ void transpose_cvt(const float* __restrict__ W, __bf16* __restrict__ WT,
                              int K, int N) {
  __shared__ float tile[32][33];
  int n0 = blockIdx.x * 32;
  int k0 = blockIdx.y * 32;
  for (int i = threadIdx.y; i < 32; i += 8)
    tile[i][threadIdx.x] = W[(size_t)(k0 + i) * N + n0 + threadIdx.x];
  __syncthreads();
  for (int i = threadIdx.y; i < 32; i += 8)
    WT[(size_t)(n0 + i) * K + k0 + threadIdx.x] = f2bf(tile[threadIdx.x][i]);
}

// ---------------------------------------------------------------- GEMM
// C = (A[M,K] @ BT[N,K]^T + bias) * alpha.  One wave per 32(M)x64(N) tile:
// two A fragments share each B fragment (8 WMMAs per 192B/lane loaded).
// MODE 0: fp32 row-major out.  MODE 1: bf16 [b][h][s][dk].  MODE 2: bf16 [b][h][dk][s].
template <int MODE>
__global__ void __launch_bounds__(128)
gemm_wmma(const __bf16* __restrict__ A, const __bf16* __restrict__ BT,
          const float* __restrict__ bias, float alpha, void* __restrict__ out,
          int Ndim, int Kdim) {
  const int lane = threadIdx.x & 31;
  const int wave = threadIdx.x >> 5;
  const int hl   = lane >> 4;   // which 16-lane half
  const int l16  = lane & 15;
  const int m0   = blockIdx.x * 32;
  const int n0   = blockIdx.y * 256 + wave * 64;

  v8f acc[2][4];
  for (int mi = 0; mi < 2; ++mi)
    for (int t = 0; t < 4; ++t)
      for (int r = 0; r < 8; ++r) acc[mi][t][r] = 0.0f;

  // A fragment: lane half 0 -> K 0..7 / 16..23, half 1 -> K 8..15 / 24..31
  const __bf16* arow0 = A + (size_t)(m0 + l16) * Kdim + hl * 8;
  const __bf16* arow1 = arow0 + (size_t)16 * Kdim;
  const __bf16* brow0 = BT + (size_t)(n0 + l16) * Kdim + hl * 8;

  for (int k0 = 0; k0 < Kdim; k0 += 32) {
    Frag a0, a1;
    a0.q[0] = *(const uint4*)(arow0 + k0);
    a0.q[1] = *(const uint4*)(arow0 + k0 + 16);
    a1.q[0] = *(const uint4*)(arow1 + k0);
    a1.q[1] = *(const uint4*)(arow1 + k0 + 16);
#pragma unroll
    for (int t = 0; t < 4; ++t) {
      Frag b;
      const __bf16* br = brow0 + (size_t)(t * 16) * Kdim + k0;
      b.q[0] = *(const uint4*)(br);
      b.q[1] = *(const uint4*)(br + 16);
      acc[0][t] = wmma_bf16(a0.v, b.v, acc[0][t]);
      acc[1][t] = wmma_bf16(a1.v, b.v, acc[1][t]);
    }
  }

#pragma unroll
  for (int mi = 0; mi < 2; ++mi) {
#pragma unroll
    for (int t = 0; t < 4; ++t) {
      const int gn = n0 + t * 16 + l16;
      const float bv = bias ? bias[gn] : 0.0f;
#pragma unroll
      for (int r = 0; r < 8; ++r) {
        const int gm = m0 + mi * 16 + r + hl * 8;
        const float val = (acc[mi][t][r] + bv) * alpha;
        if (MODE == 0) {
          ((float*)out)[(size_t)gm * Ndim + gn] = val;
        } else {
          const int bb = gm / S_, s = gm % S_;
          const int h = gn / DK_, dk = gn % DK_;
          if (MODE == 1)
            ((__bf16*)out)[(((size_t)(bb * H_ + h)) * S_ + s) * DK_ + dk] = f2bf(val);
          else
            ((__bf16*)out)[(((size_t)(bb * H_ + h)) * DK_ + dk) * S_ + s] = f2bf(val);
        }
      }
    }
  }
}

// ---------------------------------------------------------------- attention
// Q,K: bf16 [b][h][s][dk] (Q pre-scaled by 1/sqrt(DK)); VT: bf16 [b][h][dk][s].
// ctx out: bf16 [b][s][h*DK+dk].  One wave per 16-query tile, causal flash loop
// over 32-key tiles with online softmax.
__global__ void __launch_bounds__(128)
attn_flash(const __bf16* __restrict__ Q, const __bf16* __restrict__ Km,
           const __bf16* __restrict__ VT, __bf16* __restrict__ ctx) {
  __shared__ __bf16 Plds[4][16][40];   // per-wave 16x32 P tile, 80B row pitch

  const int lane = threadIdx.x & 31;
  const int wave = threadIdx.x >> 5;
  const int hl   = lane >> 4;
  const int l16  = lane & 15;
  const int bh   = blockIdx.y;               // b*H + h
  const int bb   = bh / H_, h = bh % H_;
  const int q0   = (blockIdx.x * 4 + wave) * 16;

  const size_t qkBase = (size_t)bh * S_ * DK_;

  // Q fragments: a0 covers dk 0..31, a1 covers dk 32..63
  const __bf16* qrow = Q + qkBase + (size_t)(q0 + l16) * DK_ + hl * 8;
  Frag a0, a1;
  a0.q[0] = *(const uint4*)(qrow + 0);
  a0.q[1] = *(const uint4*)(qrow + 16);
  a1.q[0] = *(const uint4*)(qrow + 32);
  a1.q[1] = *(const uint4*)(qrow + 48);

  v8f o[4];
  for (int t = 0; t < 4; ++t)
    for (int r = 0; r < 8; ++r) o[t][r] = 0.0f;
  float mi[8], li[8];
  for (int r = 0; r < 8; ++r) { mi[r] = -3.0e38f; li[r] = 0.0f; }

  const int nkt = (q0 + 16 + 31) / 32;       // causal tile count
  for (int kt = 0; kt < nkt; ++kt) {
    const int kb = kt * 32;

    // S = Q @ K^T for 32 keys: two 16x16 f32 tiles
    v8f s0, s1;
    for (int r = 0; r < 8; ++r) { s0[r] = 0.0f; s1[r] = 0.0f; }
    const __bf16* krow0 = Km + qkBase + (size_t)(kb + l16) * DK_ + hl * 8;
    const __bf16* krow1 = krow0 + 16 * DK_;
    Frag b;
    b.q[0] = *(const uint4*)(krow0);      b.q[1] = *(const uint4*)(krow0 + 16);
    s0 = wmma_bf16(a0.v, b.v, s0);
    b.q[0] = *(const uint4*)(krow0 + 32); b.q[1] = *(const uint4*)(krow0 + 48);
    s0 = wmma_bf16(a1.v, b.v, s0);
    b.q[0] = *(const uint4*)(krow1);      b.q[1] = *(const uint4*)(krow1 + 16);
    s1 = wmma_bf16(a0.v, b.v, s1);
    b.q[0] = *(const uint4*)(krow1 + 32); b.q[1] = *(const uint4*)(krow1 + 48);
    s1 = wmma_bf16(a1.v, b.v, s1);

    // online softmax (causal mask applied analytically), write P to LDS
    float c[8];
#pragma unroll
    for (int r = 0; r < 8; ++r) {
      const int qg = q0 + r + hl * 8;
      const float v0 = ((kb + l16)      <= qg) ? s0[r] : -3.0e38f;
      const float v1 = ((kb + 16 + l16) <= qg) ? s1[r] : -3.0e38f;
      const float rmax = halfmax(fmaxf(v0, v1));
      const float mnew = fmaxf(mi[r], rmax);
      const float cr = __expf(mi[r] - mnew);
      const float p0 = __expf(v0 - mnew);
      const float p1 = __expf(v1 - mnew);
      li[r] = li[r] * cr + halfsum(p0 + p1);
      mi[r] = mnew;
      c[r]  = cr;
      Plds[wave][r + hl * 8][l16]      = f2bf(p0);
      Plds[wave][r + hl * 8][16 + l16] = f2bf(p1);
    }

    // wave-local LDS sync (LDS ops are in-order per wave; wait for writes)
    asm volatile("s_wait_dscnt 0" ::: "memory");

    // reload P as an A fragment (16x32 bf16)
    Frag pa;
    const __bf16* prow = &Plds[wave][l16][hl * 8];
    pa.q[0] = *(const uint4*)(prow);
    pa.q[1] = *(const uint4*)(prow + 16);

    // rescale accumulators, then O += P @ V
#pragma unroll
    for (int t = 0; t < 4; ++t)
      for (int r = 0; r < 8; ++r) o[t][r] *= c[r];

    const __bf16* vbase = VT + (size_t)bh * DK_ * S_ + kb + hl * 8;
#pragma unroll
    for (int t = 0; t < 4; ++t) {
      Frag vb;
      const __bf16* vrow = vbase + (size_t)(t * 16 + l16) * S_;
      vb.q[0] = *(const uint4*)(vrow);
      vb.q[1] = *(const uint4*)(vrow + 16);
      o[t] = wmma_bf16(pa.v, vb.v, o[t]);
    }

    asm volatile("s_wait_dscnt 0" ::: "memory");  // reads done before next writes
  }

  // normalize and write ctx as bf16 [b][s][h*DK+dk]
#pragma unroll
  for (int t = 0; t < 4; ++t) {
#pragma unroll
    for (int r = 0; r < 8; ++r) {
      const int s  = q0 + r + hl * 8;
      const int dk = t * 16 + l16;
      const float val = o[t][r] / li[r];
      ctx[((size_t)(bb * S_ + s)) * D_ + h * DK_ + dk] = f2bf(val);
    }
  }
}

// ---------------------------------------------------------------- launch
extern "C" void kernel_launch(void* const* d_in, const int* in_sizes, int n_in,
                              void* d_out, int out_size, void* d_ws, size_t ws_size,
                              hipStream_t stream) {
  const float* x  = (const float*)d_in[0];
  // d_in[1] = causal mask (recomputed analytically; ignored)
  const float* Wq = (const float*)d_in[2];
  const float* bq = (const float*)d_in[3];
  const float* Wk = (const float*)d_in[4];
  const float* bk = (const float*)d_in[5];
  const float* Wv = (const float*)d_in[6];
  const float* bv = (const float*)d_in[7];
  const float* Wo = (const float*)d_in[8];
  const float* bo = (const float*)d_in[9];

  char* ws = (char*)d_ws;
  size_t off = 0;
  auto alloc = [&](size_t bytes) {
    void* p = ws + off;
    off += (bytes + 255) & ~(size_t)255;
    return p;
  };
  const size_t actB = (size_t)M_ * D_ * sizeof(__bf16);   // 8 MB
  const size_t wB   = (size_t)D_ * D_ * sizeof(__bf16);   // 2 MB
  __bf16* xb  = (__bf16*)alloc(actB);
  __bf16* WqT = (__bf16*)alloc(wB);
  __bf16* WkT = (__bf16*)alloc(wB);
  __bf16* WvT = (__bf16*)alloc(wB);
  __bf16* WoT = (__bf16*)alloc(wB);
  __bf16* Qb  = (__bf16*)alloc(actB);   // [b][h][s][dk], pre-scaled
  __bf16* Kb  = (__bf16*)alloc(actB);   // [b][h][s][dk]
  __bf16* VTb = (__bf16*)alloc(actB);   // [b][h][dk][s]
  __bf16* ctx = (__bf16*)alloc(actB);   // [b][s][d]
  (void)in_sizes; (void)n_in; (void)out_size; (void)ws_size;

  // 1) convert activations + weights to bf16 (weights transposed)
  {
    int n = M_ * D_;
    cvt_f32_bf16<<<n / 256, 256, 0, stream>>>(x, xb, n);
    dim3 g(D_ / 32, D_ / 32), blk(32, 8);
    transpose_cvt<<<g, blk, 0, stream>>>(Wq, WqT, D_, D_);
    transpose_cvt<<<g, blk, 0, stream>>>(Wk, WkT, D_, D_);
    transpose_cvt<<<g, blk, 0, stream>>>(Wv, WvT, D_, D_);
    transpose_cvt<<<g, blk, 0, stream>>>(Wo, WoT, D_, D_);
  }

  // 2) projections (WMMA GEMMs). Q carries the 1/sqrt(DK) scale.
  {
    dim3 g(M_ / 32, D_ / 256), blk(128);
    const float scale = 0.125f;   // 1/sqrt(64)
    gemm_wmma<1><<<g, blk, 0, stream>>>(xb, WqT, bq, scale, Qb, D_, D_);
    gemm_wmma<1><<<g, blk, 0, stream>>>(xb, WkT, bk, 1.0f,  Kb, D_, D_);
    gemm_wmma<2><<<g, blk, 0, stream>>>(xb, WvT, bv, 1.0f,  VTb, D_, D_);
  }

  // 3) causal flash attention
  {
    dim3 g(S_ / 64, B_ * H_), blk(128);
    attn_flash<<<g, blk, 0, stream>>>(Qb, Kb, VTb, ctx);
  }

  // 4) output projection, fp32 result
  {
    dim3 g(M_ / 32, D_ / 256), blk(128);
    gemm_wmma<0><<<g, blk, 0, stream>>>(ctx, WoT, bo, 1.0f, d_out, D_, D_);
  }
}
